// Model_13829794693476
// MI455X (gfx1250) — matrix operand order repaired and checked by reading, not compiled, not run on hardware
//
#include <hip/hip_runtime.h>
#include <hip/hip_bf16.h>

// ---------------- model constants ----------------
constexpr int NH  = 256;      // hidden
constexpr int OPE = 4;
constexpr int NC  = 10000;
constexpr int EC  = 120000;
constexpr int NT  = 16;
constexpr int ET  = 64;
constexpr int CF  = 64;
constexpr int CE  = 16;
constexpr int TF  = 32;
constexpr int TE  = 16;
constexpr int TG  = 8;
constexpr int GC  = 100;
constexpr int IN_C0 = CF + OPE + TG * NH + CE;  // 2132
constexpr int K0A   = CF + OPE;                  // 68 (cfeats + op_emb rows of cW0)
constexpr int W0_TFLAT_ROW = K0A;                // 68
constexpr int W0_BOT_ROW   = IN_C0 - CE;         // 2116
constexpr int IN_CR = NH + CE;                   // 272

// CDNA5 WMMA / TDM vector types (wave32)
typedef __bf16        v16bf __attribute__((ext_vector_type(16)));
typedef float         v8f   __attribute__((ext_vector_type(8)));
typedef unsigned int  v4u   __attribute__((ext_vector_type(4)));
typedef int           v8i   __attribute__((ext_vector_type(8)));
typedef int           v4i   __attribute__((ext_vector_type(4)));

// ---------------- edge preprocessing ----------------
__global__ __launch_bounds__(256) void prep_edges(const float* __restrict__ ef,
                                                  const int* __restrict__ src,
                                                  const int* __restrict__ dst,
                                                  int* __restrict__ degout,
                                                  int* __restrict__ degin,
                                                  float* __restrict__ Esum,
                                                  float* __restrict__ EsumR) {
  int tid = threadIdx.x;
  int e = blockIdx.x * 16 + (tid >> 4);
  int k = tid & 15;
  if (e >= EC) return;
  int s = src[e], d = dst[e];
  if (k == 0) {
    atomicAdd(&degout[s], 1);
    atomicAdd(&degin[d], 1);
  }
  float v = ef[e * CE + k];
  __hip_atomic_fetch_add(&Esum[d * CE + k], v, __ATOMIC_RELAXED, __HIP_MEMORY_SCOPE_AGENT);
  __hip_atomic_fetch_add(&EsumR[d * CE + k], fmaxf(v, 0.f), __ATOMIC_RELAXED, __HIP_MEMORY_SCOPE_AGENT);
}

__global__ __launch_bounds__(256) void make_norms(const int* __restrict__ degout,
                                                  const int* __restrict__ degin,
                                                  float* __restrict__ outn,
                                                  float* __restrict__ inn) {
  int n = blockIdx.x * blockDim.x + threadIdx.x;
  if (n >= NC) return;
  outn[n] = rsqrtf(fmaxf((float)degout[n], 1.f));
  inn[n]  = rsqrtf(fmaxf((float)degin[n], 1.f));
}

// ---------------- tiny device-topology GNN (single workgroup) ----------------
__global__ __launch_bounds__(256) void tgnn_kernel(const float* __restrict__ tfeats,
                                                   const float* __restrict__ tef,
                                                   const float* __restrict__ tW0,
                                                   const float* __restrict__ tb0,
                                                   const float* __restrict__ tWr,
                                                   const float* __restrict__ tbr,
                                                   const int* __restrict__ t_src,
                                                   const int* __restrict__ t_dst,
                                                   const int* __restrict__ tgroups,
                                                   const float* __restrict__ cW0,
                                                   float* __restrict__ v_out) {
  __shared__ float sx[NT][NH];
  __shared__ float sxs[NT][NH];
  __shared__ float sh[NT][IN_CR];
  __shared__ float son[NT], sin_[NT];
  __shared__ int   sdo[NT], sdi[NT];
  int tid = threadIdx.x;

  if (tid < NT) { sdo[tid] = 0; sdi[tid] = 0; }
  __syncthreads();
  if (tid < ET) { atomicAdd(&sdo[t_src[tid]], 1); atomicAdd(&sdi[t_dst[tid]], 1); }
  __syncthreads();
  if (tid < NT) {
    son[tid]  = rsqrtf(fmaxf((float)sdo[tid], 1.f));
    sin_[tid] = rsqrtf(fmaxf((float)sdi[tid], 1.f));
  }
  __syncthreads();

  // ---- layer 0: F=32, double_act, relu_out, no residual ----
  for (int idx = tid; idx < NT * TF; idx += 256) {
    int n = idx / TF, k = idx % TF;
    sxs[n][k] = tfeats[idx] * son[n];
  }
  for (int idx = tid; idx < NT * IN_CR; idx += 256) ((float*)sh)[idx] = 0.f;
  __syncthreads();
  for (int idx = tid; idx < ET * TF; idx += 256) {
    int e = idx / TF, k = idx % TF;
    atomicAdd(&sh[t_dst[e]][k], fmaxf(sxs[t_src[e]][k], 0.f));
  }
  for (int idx = tid; idx < ET * TE; idx += 256) {
    int e = idx / TE, k = idx % TE;
    atomicAdd(&sh[t_dst[e]][TF + k], fmaxf(tef[e * TE + k], 0.f));
  }
  __syncthreads();
  for (int idx = tid; idx < NT * NH; idx += 256) {
    int n = idx / NH, j = idx % NH;
    float s = 0.f;
    for (int k = 0; k < TF + TE; k++) s += sh[n][k] * tW0[k * NH + j];
    s = s * sin_[n] + tb0[j];
    sx[n][j] = fmaxf(s, 0.f);
  }
  __syncthreads();

  // ---- layers 1..3: F=256 ----
  for (int L = 0; L < 3; L++) {
    const float* W = tWr + (size_t)L * IN_CR * NH;
    const float* b = tbr + L * NH;
    bool resi = (L < 2), relu = (L < 2);
    for (int idx = tid; idx < NT * NH; idx += 256) {
      int n = idx / NH, k = idx % NH;
      sxs[n][k] = sx[n][k] * son[n];
    }
    for (int idx = tid; idx < NT * IN_CR; idx += 256) ((float*)sh)[idx] = 0.f;
    __syncthreads();
    for (int idx = tid; idx < ET * NH; idx += 256) {
      int e = idx / NH, k = idx % NH;
      atomicAdd(&sh[t_dst[e]][k], sxs[t_src[e]][k]);
    }
    for (int idx = tid; idx < ET * TE; idx += 256) {
      int e = idx / TE, k = idx % TE;
      atomicAdd(&sh[t_dst[e]][NH + k], tef[e * TE + k]);
    }
    __syncthreads();
    for (int t = 0; t < NT; t++) {
      int n = t, j = tid;
      float s = 0.f;
      for (int k = 0; k < IN_CR; k++) s += sh[n][k] * W[k * NH + j];
      s = s * sin_[n] + b[j];
      if (relu) s = fmaxf(s, 0.f);
      if (resi) s += sxs[n][j];
      sx[n][j] = s;
    }
    __syncthreads();
  }

  // ---- t_emb pairs -> relu(t_flat) -> v = relu(t_flat) @ cW0[68:2116] ----
  float* tflat = (float*)sh;
  for (int idx = tid; idx < TG * NH; idx += 256) {
    int g = idx >> 8, j = idx & (NH - 1);
    float tv = sx[tgroups[2 * g]][j] + sx[tgroups[2 * g + 1]][j];
    tflat[idx] = fmaxf(tv, 0.f);
  }
  __syncthreads();
  {
    int j = tid;
    float s = 0.f;
    for (int k = 0; k < TG * NH; k++) s += tflat[k] * cW0[(size_t)(W0_TFLAT_ROW + k) * NH + j];
    v_out[j] = s;
  }
}

// ---------------- WMMA GEMM, layers 1-5 ----------------
// Y[n, :] = out_norm[n] * (feat[n, :256] @ W_top[256,256])
// A tile (16 rows x 256 K, f32, 16 KB) staged into LDS once per block by the
// Tensor Data Mover; all 8 waves build A fragments from LDS (ds_load), each
// wave owns one 16x16 output tile (8 x v_wmma_f32_16x16x32_bf16 per tile).
__global__ __launch_bounds__(256) void gemm_node(const float* __restrict__ X,
                                                 const float* __restrict__ W,
                                                 const float* __restrict__ outn,
                                                 float* __restrict__ Y) {
  __shared__ float sA[16 * NH] __attribute__((aligned(16)));   // 16 KB A tile

  int wave = threadIdx.x >> 5;
  int lane = threadIdx.x & 31;
  int r0 = blockIdx.x * 16;
  int c0 = blockIdx.y * 128 + wave * 16;
  int col = c0 + (lane & 15);
  int hi  = lane >> 4;           // half-wave select
  int kA  = hi * 8;              // A: K sub-block per half-wave
  int kB  = hi * 16;             // unused; B loads also use kA

  // ---- TDM: DMA the 16x256 f32 tile X[r0:r0+16, :] into LDS (wave 0 only) ----
  if (wave == 0) {
    unsigned lds_off = (unsigned)(unsigned long long)(const void*)sA;  // LDS aperture: low 32 bits = LDS offset
    unsigned long long ga = (unsigned long long)(const void*)(X + (size_t)r0 * NH);
    // D# group 0: count=1 | lds_addr | global_addr[56:0] | type=2
    v4u g0 = { 1u,
               lds_off,
               (unsigned)(ga & 0xFFFFFFFFu),
               (unsigned)((ga >> 32) & 0x01FFFFFFu) | (2u << 30) };
    // D# group 1: data_size=4B(code 2); tensor 256x16; tile 256x16; stride0=256
    v8i g1 = { (int)(2u << 16),     // workgroup_mask=0, data_size=2
               (int)(256u << 16),   // atomic_barrier_addr=0 | tensor_dim0.lo=256
               (int)(16u << 16),    // tensor_dim0.hi=0 | tensor_dim1.lo=16
               (int)(256u << 16),   // tensor_dim1.hi=0 | tile_dim0=256
               16,                  // tile_dim1=16 | tile_dim2=0
               256,                 // tensor_dim0_stride.lo=256
               0,                   // stride0.hi=0 | tensor_dim1_stride.lo=0
               0 };
    v4i gz4 = { 0, 0, 0, 0 };
    v8i gz8 = { 0, 0, 0, 0, 0, 0, 0, 0 };
    __builtin_amdgcn_tensor_load_to_lds(g0, g1, gz4, gz4, gz8, 0);
    __builtin_amdgcn_s_wait_tensorcnt(0);
  }
  __syncthreads();

  v8f c;
#pragma unroll
  for (int i = 0; i < 8; i++) c[i] = 0.f;

  const float* Arow = sA + (size_t)(lane & 15) * NH;
#pragma unroll
  for (int k0 = 0; k0 < NH; k0 += 32) {
    v16bf a, b;
#pragma unroll
    for (int e = 0; e < 8; e++) a[e] = (__bf16)Arow[k0 + kA + e];
#pragma unroll
    for (int e = 0; e < 8; e++) a[8 + e] = (__bf16)Arow[k0 + 16 + kA + e];
#pragma unroll
    for (int e = 0; e < 8; e++) b[e] = (__bf16)W[(size_t)(k0 + kA + e) * NH + col];
#pragma unroll
    for (int e = 0; e < 8; e++) b[8 + e] = (__bf16)W[(size_t)(k0 + 16 + kA + e) * NH + col];
    c = __builtin_amdgcn_wmma_f32_16x16x32_bf16(false, a, false, b, (short)0, c, false, false);
  }

#pragma unroll
  for (int i = 0; i < 8; i++) {
    int r = r0 + i + hi * 8;
    Y[(size_t)r * NH + col] = c[i] * outn[r];
  }
}

// ---------------- Layer-0 GEMM ----------------
// Y0[n,:] = out_norm[n] * ( relu([cfeats|op_emb]) @ W0[0:68] + v )
__global__ __launch_bounds__(256) void gemm_layer0(const float* __restrict__ cfeats,
                                                   const float* __restrict__ opemb,
                                                   const int* __restrict__ ctypes,
                                                   const float* __restrict__ W0,
                                                   const float* __restrict__ vvec,
                                                   const float* __restrict__ outn,
                                                   float* __restrict__ Y) {
  int wave = threadIdx.x >> 5;
  int lane = threadIdx.x & 31;
  int r0 = blockIdx.x * 16;
  int c0 = blockIdx.y * 128 + wave * 16;
  int row = r0 + (lane & 15);
  int col = c0 + (lane & 15);
  int hi  = lane >> 4;
  int kA  = hi * 8;
  int kB  = hi * 16;

  float vc = vvec[col];
  v8f c;
#pragma unroll
  for (int i = 0; i < 8; i++) c[i] = vc;   // rank-1 t_flat contribution as C-init

  int ct = ctypes[row] * OPE;
#pragma unroll
  for (int k0 = 0; k0 < 96; k0 += 32) {    // K=68 padded to 96
    v16bf a, b;
#pragma unroll
    for (int e = 0; e < 16; e++) {
      int kk = k0 + ((e < 8) ? (kA + e) : (16 + kA + e - 8));
      float av = 0.f;
      if (kk < CF)        av = cfeats[(size_t)row * CF + kk];
      else if (kk < K0A)  av = opemb[ct + (kk - CF)];
      a[e] = (__bf16)fmaxf(av, 0.f);       // double_act relu on node features
    }
#pragma unroll
    for (int e = 0; e < 16; e++) {
      int kk = k0 + ((e < 8) ? (kA + e) : (16 + kA + e - 8));
      float bv = (kk < K0A) ? W0[(size_t)kk * NH + col] : 0.f;
      b[e] = (__bf16)bv;
    }
    c = __builtin_amdgcn_wmma_f32_16x16x32_bf16(false, a, false, b, (short)0, c, false, false);
  }

#pragma unroll
  for (int i = 0; i < 8; i++) {
    int r = r0 + i + hi * 8;
    Y[(size_t)r * NH + col] = c[i] * outn[r];
  }
}

// ---------------- edge scatter: acc[dst] += Y[src]  (one wave per edge) ----------------
__global__ __launch_bounds__(256) void scatter_edges(const float* __restrict__ Y,
                                                     const int* __restrict__ src,
                                                     const int* __restrict__ dst,
                                                     float* __restrict__ acc) {
  int e = blockIdx.x * 8 + (threadIdx.x >> 5);
  int lane = threadIdx.x & 31;
  if (e >= EC) return;
  size_t s = (size_t)src[e] * NH;
  size_t d = (size_t)dst[e] * NH;
#pragma unroll
  for (int i = 0; i < 8; i++) {
    float val = Y[s + lane + i * 32];
    __hip_atomic_fetch_add(&acc[d + lane + i * 32], val, __ATOMIC_RELAXED, __HIP_MEMORY_SCOPE_AGENT);
  }
}

// ---------------- per-layer epilogue ----------------
__global__ __launch_bounds__(256) void finalize(const float* __restrict__ acc,
                                                const float* __restrict__ Esum16,
                                                const float* __restrict__ Wbot,   // [16,256]
                                                const float* __restrict__ bias,
                                                const float* __restrict__ inn,
                                                const float* __restrict__ outn,
                                                const float* __restrict__ featPrev,
                                                float* __restrict__ featOut,
                                                int flags) {                      // 1=relu, 2=residual
  int n = blockIdx.x, j = threadIdx.x;
  float s = acc[(size_t)n * NH + j];
  const float* es = Esum16 + n * CE;
#pragma unroll
  for (int k = 0; k < CE; k++) s += es[k] * Wbot[k * NH + j];
  s = s * inn[n] + bias[j];
  if (flags & 1) s = fmaxf(s, 0.f);
  if (flags & 2) s += outn[n] * featPrev[(size_t)n * NH + j];
  featOut[(size_t)n * NH + j] = s;
}

// ---------------- heads: group-sum + log_sigmoid + log_softmax ----------------
__global__ __launch_bounds__(256) void head_kernel(const float* __restrict__ feat,
                                                   const int* __restrict__ cgroups,
                                                   const float* __restrict__ sW,
                                                   const float* __restrict__ sb,
                                                   const float* __restrict__ nW,
                                                   const float* __restrict__ nb,
                                                   float* __restrict__ out) {
  __shared__ float sc[NH];
  __shared__ float red[NH];
  __shared__ float results[16];
  int g = blockIdx.x, j = threadIdx.x;
  float s = 0.f;
  for (int i = 0; i < NC / GC; i++) s += feat[(size_t)cgroups[g * (NC / GC) + i] * NH + j];
  sc[j] = s;
  __syncthreads();
  for (int o = 0; o < 1 + TG; o++) {
    float p = (o == 0) ? sc[j] * nW[j] : sc[j] * sW[j * TG + (o - 1)];
    red[j] = p;
    __syncthreads();
    for (int st = 128; st > 0; st >>= 1) {
      if (j < st) red[j] += red[j + st];
      __syncthreads();
    }
    if (j == 0) results[o] = red[0] + ((o == 0) ? nb[0] : sb[o - 1]);
    __syncthreads();
  }
  if (j == 0) {
    float zn = results[0];
    out[g * 9] = fminf(zn, 0.f) - log1pf(expf(-fabsf(zn)));   // log_sigmoid
    float m = results[1];
    for (int k = 2; k <= TG; k++) m = fmaxf(m, results[k]);
    float se = 0.f;
    for (int k = 1; k <= TG; k++) se += expf(results[k] - m);
    float lse = m + logf(se);
    for (int k = 1; k <= TG; k++) out[g * 9 + k] = results[k] - lse;
  }
}

// ---------------- driver ----------------
extern "C" void kernel_launch(void* const* d_in, const int* in_sizes, int n_in,
                              void* d_out, int out_size, void* d_ws, size_t ws_size,
                              hipStream_t stream) {
  const float* cfeats = (const float*)d_in[0];
  const float* cedge  = (const float*)d_in[1];
  const float* tfeats = (const float*)d_in[2];
  const float* tedge  = (const float*)d_in[3];
  const float* opemb  = (const float*)d_in[4];
  const float* tW0    = (const float*)d_in[5];
  const float* tb0    = (const float*)d_in[6];
  const float* tWr    = (const float*)d_in[7];
  const float* tbr    = (const float*)d_in[8];
  const float* cW0    = (const float*)d_in[9];
  const float* cb0    = (const float*)d_in[10];
  const float* cWr    = (const float*)d_in[11];
  const float* cbr    = (const float*)d_in[12];
  const float* sW     = (const float*)d_in[13];
  const float* sb     = (const float*)d_in[14];
  const float* nW     = (const float*)d_in[15];
  const float* nb     = (const float*)d_in[16];
  const int* ctypes   = (const int*)d_in[17];
  const int* c_src    = (const int*)d_in[18];
  const int* c_dst    = (const int*)d_in[19];
  const int* t_src    = (const int*)d_in[20];
  const int* t_dst    = (const int*)d_in[21];
  const int* cgroups  = (const int*)d_in[22];
  const int* tgroups  = (const int*)d_in[23];

  char* ws = (char*)d_ws;
  size_t off = 0;
  auto take = [&](size_t bytes) -> void* {
    void* p = ws + off;
    off += (bytes + 255) & ~(size_t)255;
    return p;
  };
  int*   degout = (int*)take(NC * 4);
  int*   degin  = (int*)take(NC * 4);
  float* Esum   = (float*)take((size_t)NC * CE * 4);
  float* EsumR  = (float*)take((size_t)NC * CE * 4);
  size_t zero_bytes = off;
  float* outn = (float*)take(NC * 4);
  float* inn  = (float*)take(NC * 4);
  float* vvec = (float*)take(NH * 4);
  float* Y    = (float*)take((size_t)NC * NH * 4);
  float* acc  = (float*)take((size_t)NC * NH * 4);
  float* fA   = (float*)take((size_t)NC * NH * 4);
  float* fB   = (float*)take((size_t)NC * NH * 4);

  (void)hipMemsetAsync(ws, 0, zero_bytes, stream);

  prep_edges<<<EC / 16, 256, 0, stream>>>(cedge, c_src, c_dst, degout, degin, Esum, EsumR);
  make_norms<<<(NC + 255) / 256, 256, 0, stream>>>(degout, degin, outn, inn);
  tgnn_kernel<<<1, 256, 0, stream>>>(tfeats, tedge, tW0, tb0, tWr, tbr, t_src, t_dst,
                                     tgroups, cW0, vvec);

  dim3 ggrid(NC / 16, NH / 128);

  // ---- c-layer 0 ----
  (void)hipMemsetAsync(acc, 0, (size_t)NC * NH * 4, stream);
  gemm_layer0<<<ggrid, 256, 0, stream>>>(cfeats, opemb, ctypes, cW0, vvec, outn, Y);
  scatter_edges<<<EC / 8, 256, 0, stream>>>(Y, c_src, c_dst, acc);
  finalize<<<NC, NH, 0, stream>>>(acc, EsumR, cW0 + (size_t)W0_BOT_ROW * NH, cb0,
                                  inn, outn, fA, fA, /*relu*/ 1);

  // ---- c-layers 1..5 ----
  float* cur = fA;
  float* nxt = fB;
  for (int i = 0; i < 5; i++) {
    const float* W  = cWr + (size_t)i * IN_CR * NH;        // W_top = rows 0..255
    const float* Wb = W + (size_t)NH * NH;                 // W_bot = rows 256..271
    const float* b  = cbr + i * NH;
    int flags = (i < 4) ? 3 : 0;                           // relu+residual for 1..4
    (void)hipMemsetAsync(acc, 0, (size_t)NC * NH * 4, stream);
    gemm_node<<<ggrid, 256, 0, stream>>>(cur, W, outn, Y);
    scatter_edges<<<EC / 8, 256, 0, stream>>>(Y, c_src, c_dst, acc);
    finalize<<<NC, NH, 0, stream>>>(acc, Esum, Wb, b, inn, outn, cur, nxt, flags);
    float* t = cur; cur = nxt; nxt = t;
  }

  head_kernel<<<GC, NH, 0, stream>>>(cur, cgroups, sW, sb, nW, nb, (float*)d_out);
}
